// ASTGCN_79946521247969
// MI455X (gfx1250) — matrix operand order repaired and checked by reading, not compiled, not run on hardware
//
#include <hip/hip_runtime.h>
#include <hip/hip_bf16.h>

// ---------------------------------------------------------------------------
// ASTGCN forward for MI455X (gfx1250, wave32).
// Large GEMMs + spatial attention on v_wmma_f32_16x16x32_bf16; all B-operand
// traffic staged through LDS with global_load_async_to_lds_b128 (ASYNCcnt),
// double-buffered in the attention kernel to overlap DMA with WMMA.
// Shapes: B=8 T=24 N=1024 F=3 H=256 NH=8 HD=32 P=12.  M = B*T*N = 196608.
// ---------------------------------------------------------------------------

typedef __bf16 bf16;
typedef __bf16 v16bf __attribute__((ext_vector_type(16)));
typedef __bf16 v8bf  __attribute__((ext_vector_type(8)));
typedef float  v8f   __attribute__((ext_vector_type(8)));

#define M_ROWS 196608      // B*T*N
#define HDIM   256
#define KDIM   256         // inner dim of every WMMA GEMM here
#define LPITCH 264         // KDIM + 8 pad elems -> 528B row pitch (bank-skewed)
#define NNODE  1024
#define TSTEP  24
#define BT     192         // B*T
#define BN     8192        // B*N
#define LDSW_BYTES (256 * LPITCH * 2)   // 135168

// spattn LDS carving (bytes)
#define KM_PITCH  264                    // 64-key chunk rows of 256 K elems
#define KM_BYTES  (64 * KM_PITCH * 2)    // 33792 per buffer
#define HT_PITCH  72                     // 256 hdim rows of 64 key elems
#define HT_BYTES  (256 * HT_PITCH * 2)   // 36864 per buffer
#define PB_OFF    (2 * KM_BYTES + 2 * HT_BYTES)           // 141312
#define SPATTN_LDS (PB_OFF + 4 * 16 * 64 * 2)             // 149504

__device__ __forceinline__ v8f v8f_zero() {
  v8f z;
#pragma unroll
  for (int e = 0; e < 8; ++e) z[e] = 0.0f;
  return z;
}

__device__ __forceinline__ v8f wmma_bf16(v16bf a, v16bf b, v8f c) {
  return __builtin_amdgcn_wmma_f32_16x16x32_bf16(false, a, false, b,
                                                 (short)0, c, false, false);
}

// A-fragment (16x32 bf16, contiguous k): p must already include +16*half elems.
__device__ __forceinline__ v16bf ld_a(const bf16* p) {
  v8bf lo = *(const v8bf*)p;
  v8bf hi = *(const v8bf*)(p + 8);
  return __builtin_shufflevector(lo, hi, 0, 1, 2, 3, 4, 5, 6, 7,
                                         8, 9, 10, 11, 12, 13, 14, 15);
}

// 16B async DMA: global -> LDS, tracked by ASYNCcnt (no VGPR round-trip)
__device__ __forceinline__ void cp16_async(void* ldsp, const void* g) {
  unsigned l = (unsigned)(uintptr_t)ldsp;
  asm volatile("global_load_async_to_lds_b128 %0, %1, off"
               :: "v"(l), "v"(g) : "memory");
}

// (b,n,t) row -> (b,t,n) row  (ht = h2.transpose fold)
__device__ __forceinline__ long remap_sp2t(long r) {
  long b   = r / (long)(NNODE * TSTEP);
  long rem = r % (long)(NNODE * TSTEP);
  long n   = rem / TSTEP;
  long t   = rem % TSTEP;
  return (b * TSTEP + t) * NNODE + n;
}

// ---------------------------------------------------------------------------
// Layer 0: h = x @ w_in.T + b_in   (K=3, bandwidth-trivial; emit bf16)
// ---------------------------------------------------------------------------
__global__ void k_in(const float* __restrict__ x, const float* __restrict__ w_in,
                     const float* __restrict__ b_in, bf16* __restrict__ h) {
  int i = blockIdx.x * blockDim.x + threadIdx.x;    // over M_ROWS*HDIM
  int r = i >> 8, c = i & 255;
  const float* xp = x + (long)r * 3;
  float acc = b_in[c] + xp[0] * w_in[c * 3 + 0] + xp[1] * w_in[c * 3 + 1] +
              xp[2] * w_in[c * 3 + 2];
  h[i] = (bf16)acc;
}

__global__ void k_cast(const float* __restrict__ s, bf16* __restrict__ d, int n) {
  int i = blockIdx.x * blockDim.x + threadIdx.x;
  if (i < n) d[i] = (bf16)s[i];
}

// hT[bt][h][n] = h[(bt*N+n)*H + h]  (for P@V B-fragments, contiguous loads)
__global__ void k_transpose(const bf16* __restrict__ h, bf16* __restrict__ hT) {
  long i = (long)blockIdx.x * blockDim.x + threadIdx.x;  // bt,c,n with n inner
  long bt  = i / (NNODE * HDIM);
  long rem = i % (NNODE * HDIM);
  long c   = rem / NNODE;
  long n   = rem % NNODE;
  hT[i] = h[(bt * NNODE + n) * HDIM + c];
}

// ---------------------------------------------------------------------------
// Generic bf16 WMMA GEMM:  C[M,Nc] = act(A[M,256] @ W[Nc,256]^T + bias)
// wave tile 32x64, block = 8 waves = 64x256.  Weight slice (128KB) async-
// staged to LDS once per workgroup, B-fragments served via DS.
// ---------------------------------------------------------------------------
template <int RELU, int REMAP>
__global__ __launch_bounds__(256) void k_gemm(const bf16* __restrict__ A,
                                              const bf16* __restrict__ W,
                                              const float* __restrict__ bias,
                                              bf16* __restrict__ C,
                                              int Nc) {
  extern __shared__ __align__(16) char smem_raw[];
  bf16* ldsW = (bf16*)smem_raw;                       // [256][LPITCH]

  const int lane = threadIdx.x & 31;
  const int wave = threadIdx.x >> 5;
  const int hl = lane >> 4;          // half of wave
  const int r  = lane & 15;
  const long mBase = (long)blockIdx.x * 64 + (wave >> 2) * 32;
  const int  nBase = blockIdx.y * 256 + (wave & 3) * 64;

  // ---- async-stage W rows [blockIdx.y*256, +256) x 256 into LDS ----
  {
    const bf16* wsrc = W + (size_t)blockIdx.y * 256 * KDIM;
#pragma unroll
    for (int it = 0; it < 32; ++it) {
      int chunk = it * 256 + threadIdx.x;      // [0, 8192)
      int row = chunk >> 5;                    // W row (output col)
      int c16 = chunk & 31;                    // 16B chunk within row
      cp16_async((char*)ldsW + row * (LPITCH * 2) + c16 * 16,
                 wsrc + row * KDIM + c16 * 8);
    }
    asm volatile("s_wait_asynccnt 0" ::: "memory");
    __syncthreads();
  }

  long row0 = mBase + r, row1 = mBase + 16 + r;
  if (REMAP) { row0 = remap_sp2t(row0); row1 = remap_sp2t(row1); }
  const bf16* a0p = A + row0 * KDIM + 16 * hl;
  const bf16* a1p = A + row1 * KDIM + 16 * hl;
  __builtin_prefetch(a0p, 0, 1);
  __builtin_prefetch(a1p, 0, 1);

  const int localn = (wave & 3) * 64;   // wave's W row base in staged slice

  v8f acc[2][4];
#pragma unroll
  for (int i = 0; i < 2; ++i)
#pragma unroll
    for (int j = 0; j < 4; ++j) acc[i][j] = v8f_zero();

#pragma unroll
  for (int k0 = 0; k0 < KDIM; k0 += 32) {
    v16bf a0 = ld_a(a0p + k0);
    v16bf a1 = ld_a(a1p + k0);
#pragma unroll
    for (int j = 0; j < 4; ++j) {
      v16bf b = *(const v16bf*)(ldsW + (localn + j * 16 + r) * LPITCH +
                                16 * hl + k0);
      acc[0][j] = wmma_bf16(a0, b, acc[0][j]);
      acc[1][j] = wmma_bf16(a1, b, acc[1][j]);
    }
  }

#pragma unroll
  for (int i = 0; i < 2; ++i)
#pragma unroll
    for (int j = 0; j < 4; ++j) {
      int col = nBase + j * 16 + r;
      float bv = bias[col];
#pragma unroll
      for (int q = 0; q < 8; ++q) {
        long rowo = mBase + i * 16 + 8 * hl + q;   // C layout: row = 8*half + vgpr
        float v = acc[i][j][q] + bv;
        if (RELU) v = fmaxf(v, 0.0f);
        C[rowo * Nc + col] = (bf16)v;
      }
    }
}

// ---------------------------------------------------------------------------
// Spatial attention (flash style), per (b,t): softmax(QK^T/16) @ H.
// block = 4 waves x 16 query rows; keys streamed in 64-chunks.  K-chunk and
// hT-chunk are double-buffer async-staged to LDS (shared by all 4 waves),
// s_wait_asynccnt 32 overlaps chunk c+1 DMA with chunk c WMMAs.
// ---------------------------------------------------------------------------
__global__ __launch_bounds__(128) void k_spattn(const bf16* __restrict__ Q,
                                                const bf16* __restrict__ Km,
                                                const bf16* __restrict__ hT,
                                                bf16* __restrict__ O) {
  extern __shared__ __align__(16) char smem[];
  const int lane = threadIdx.x & 31;
  const int wave = threadIdx.x >> 5;
  const int hl = lane >> 4;
  const int r  = lane & 15;
  const int bt = blockIdx.y;
  const int qBase = blockIdx.x * 64 + wave * 16;
  const long rowbase = (long)bt * NNODE;
  const bf16* qrow = Q + (rowbase + qBase + r) * HDIM + 16 * hl;
  const bf16* hTb  = hT + (long)bt * HDIM * NNODE;
  bf16* pb = (bf16*)(smem + PB_OFF) + wave * 16 * 64;

  // stage key chunk `key0` (64x256) and hT chunk (256x64) into buffer `buf`
  auto stage = [&](int key0, int buf) {
    char* kmdst = smem + buf * KM_BYTES;
    char* htdst = smem + 2 * KM_BYTES + buf * HT_BYTES;
#pragma unroll
    for (int it = 0; it < 16; ++it) {
      int idx = it * 128 + threadIdx.x;        // [0,2048): 64 rows x 32 chunks
      int row = idx >> 5, c16 = idx & 31;
      cp16_async(kmdst + row * (KM_PITCH * 2) + c16 * 16,
                 Km + (rowbase + key0 + row) * HDIM + c16 * 8);
    }
#pragma unroll
    for (int it = 0; it < 16; ++it) {
      int idx = it * 128 + threadIdx.x;        // [0,2048): 256 rows x 8 chunks
      int row = idx >> 3, c8 = idx & 7;
      cp16_async(htdst + row * (HT_PITCH * 2) + c8 * 16,
                 hTb + (long)row * NNODE + key0 + c8 * 8);
    }
  };

  v8f o[16];
#pragma unroll
  for (int nt = 0; nt < 16; ++nt) o[nt] = v8f_zero();
  float rmax[8], rsum[8];
#pragma unroll
  for (int j = 0; j < 8; ++j) { rmax[j] = -1e30f; rsum[j] = 0.0f; }

  const float scale = 0.0625f;  // 1/sqrt(256)

  stage(0, 0);                  // prologue DMA

  for (int c = 0; c < 16; ++c) {   // 16 key chunks of 64
    if (c + 1 < 16) {
      stage((c + 1) * 64, (c + 1) & 1);
      asm volatile("s_wait_asynccnt 32" ::: "memory");  // chunk c landed
    } else {
      asm volatile("s_wait_asynccnt 0" ::: "memory");
    }
    __syncthreads();
    const bf16* kmc = (const bf16*)(smem + (c & 1) * KM_BYTES);
    const bf16* htc = (const bf16*)(smem + 2 * KM_BYTES + (c & 1) * HT_BYTES);

    v8f s[4];
#pragma unroll
    for (int kt = 0; kt < 4; ++kt) s[kt] = v8f_zero();

#pragma unroll
    for (int k0 = 0; k0 < HDIM; k0 += 32) {
      v16bf a = ld_a(qrow + k0);
#pragma unroll
      for (int kt = 0; kt < 4; ++kt) {
        v16bf b = *(const v16bf*)(kmc + (kt * 16 + r) * KM_PITCH + 16 * hl + k0);
        s[kt] = wmma_bf16(a, b, s[kt]);
      }
    }

    // ---- online softmax over this chunk (rows = 8*hl + j) ----
#pragma unroll
    for (int kt = 0; kt < 4; ++kt)
#pragma unroll
      for (int j = 0; j < 8; ++j) s[kt][j] *= scale;

#pragma unroll
    for (int j = 0; j < 8; ++j) {
      float m = fmaxf(fmaxf(s[0][j], s[1][j]), fmaxf(s[2][j], s[3][j]));
#pragma unroll
      for (int msk = 1; msk < 16; msk <<= 1) m = fmaxf(m, __shfl_xor(m, msk, 32));
      float nmax = fmaxf(rmax[j], m);
      float alpha = __expf(rmax[j] - nmax);
      rmax[j] = nmax;
      rsum[j] *= alpha;
#pragma unroll
      for (int nt = 0; nt < 16; ++nt) o[nt][j] *= alpha;
    }

#pragma unroll
    for (int kt = 0; kt < 4; ++kt)
#pragma unroll
      for (int j = 0; j < 8; ++j) s[kt][j] = __expf(s[kt][j] - rmax[j]);

#pragma unroll
    for (int j = 0; j < 8; ++j) {
      float ss = s[0][j] + s[1][j] + s[2][j] + s[3][j];
#pragma unroll
      for (int msk = 1; msk < 16; msk <<= 1) ss += __shfl_xor(ss, msk, 32);
      rsum[j] += ss;
    }

    // ---- P (C-layout) -> LDS patch -> A-fragments ----
#pragma unroll
    for (int kt = 0; kt < 4; ++kt)
#pragma unroll
      for (int j = 0; j < 8; ++j)
        pb[(8 * hl + j) * 64 + kt * 16 + r] = (bf16)s[kt][j];
    asm volatile("s_wait_dscnt 0" ::: "memory");

    // ---- O += P @ Hchunk  (B-fragments from staged hT) ----
#pragma unroll
    for (int ks = 0; ks < 2; ++ks) {
      v16bf pa = ld_a(pb + r * 64 + ks * 32 + 16 * hl);
#pragma unroll
      for (int nt = 0; nt < 16; ++nt) {
        v16bf hb = *(const v16bf*)(htc + (nt * 16 + r) * HT_PITCH + ks * 32 +
                                   16 * hl);
        o[nt] = wmma_bf16(pa, hb, o[nt]);
      }
    }
    __syncthreads();   // protect buffer c&1 before its next DMA overwrite
  }

  // epilogue: normalize + store bf16
#pragma unroll
  for (int nt = 0; nt < 16; ++nt) {
    int col = nt * 16 + r;
#pragma unroll
    for (int j = 0; j < 8; ++j) {
      long row = rowbase + qBase + 8 * hl + j;
      O[row * HDIM + col] = (bf16)(o[nt][j] * (1.0f / rsum[j]));
    }
  }
}

// ---------------------------------------------------------------------------
// Temporal attention: per (b*n, head, t_query).  T=24, HD=32 — VALU is fine.
// qkv rows = (bn*T + t), cols 768 = [q | k | v].
// ---------------------------------------------------------------------------
__global__ void k_tattn(const bf16* __restrict__ qkv, bf16* __restrict__ o) {
  int i = blockIdx.x * blockDim.x + threadIdx.x;  // over BN*NH*T
  int tq = i % TSTEP;
  int rem = i / TSTEP;
  int head = rem & 7;
  long bn = rem >> 3;
  const bf16* base = qkv + bn * TSTEP * 768;
  const bf16* qr = base + (long)tq * 768 + head * 32;
  float qv[32];
#pragma unroll
  for (int d = 0; d < 32; ++d) qv[d] = (float)qr[d];

  float sc[24], mx = -1e30f;
  for (int tk = 0; tk < TSTEP; ++tk) {
    const bf16* kr = base + (long)tk * 768 + 256 + head * 32;
    float s = 0.0f;
#pragma unroll
    for (int d = 0; d < 32; ++d) s += qv[d] * (float)kr[d];
    s *= 0.17677669529663687f;  // 1/sqrt(32)
    sc[tk] = s;
    mx = fmaxf(mx, s);
  }
  float sum = 0.0f;
  for (int tk = 0; tk < TSTEP; ++tk) { sc[tk] = __expf(sc[tk] - mx); sum += sc[tk]; }
  float inv = 1.0f / sum;

  float acc[32];
#pragma unroll
  for (int d = 0; d < 32; ++d) acc[d] = 0.0f;
  for (int tk = 0; tk < TSTEP; ++tk) {
    float p = sc[tk] * inv;
    const bf16* vr = base + (long)tk * 768 + 512 + head * 32;
#pragma unroll
    for (int d = 0; d < 32; ++d) acc[d] += p * (float)vr[d];
  }
  bf16* orow = o + (bn * TSTEP + tq) * HDIM + head * 32;
#pragma unroll
  for (int d = 0; d < 32; ++d) orow[d] = (bf16)acc[d];
}

// ---------------------------------------------------------------------------
// Final: take t=T-1 rows of g2 (still in (b,n,t) order), project to 36 dims,
// scatter to out[b,p,n,f].
// ---------------------------------------------------------------------------
__global__ void k_final(const bf16* __restrict__ g2, const float* __restrict__ w_out,
                        const float* __restrict__ b_out, float* __restrict__ out) {
  int i = blockIdx.x * blockDim.x + threadIdx.x;  // over BN*36
  int j = i % 36;
  long bn = i / 36;
  long b = bn >> 10, n = bn & 1023;
  const bf16* hrow = g2 + (bn * TSTEP + (TSTEP - 1)) * HDIM;
  const float* wrow = w_out + (long)j * HDIM;
  float acc = b_out[j];
  for (int k = 0; k < HDIM; ++k) acc += (float)hrow[k] * wrow[k];
  int p = j / 3, f = j % 3;
  out[((b * 12 + p) * NNODE + n) * 3 + f] = acc;
}

// ---------------------------------------------------------------------------
extern "C" void kernel_launch(void* const* d_in, const int* in_sizes, int n_in,
                              void* d_out, int out_size, void* d_ws, size_t ws_size,
                              hipStream_t stream) {
  const float* x     = (const float*)d_in[0];
  const float* w_in  = (const float*)d_in[2];
  const float* b_in  = (const float*)d_in[3];
  const float* w_s1  = (const float*)d_in[4];
  const float* b_s1  = (const float*)d_in[5];
  const float* w_s2  = (const float*)d_in[6];
  const float* b_s2  = (const float*)d_in[7];
  const float* w_qkv = (const float*)d_in[8];
  const float* b_qkv = (const float*)d_in[9];
  const float* w_o   = (const float*)d_in[10];
  const float* b_o   = (const float*)d_in[11];
  const float* w_g1  = (const float*)d_in[12];
  const float* b_g1  = (const float*)d_in[13];
  const float* w_g2  = (const float*)d_in[14];
  const float* b_g2  = (const float*)d_in[15];
  const float* w_out = (const float*)d_in[16];
  const float* b_out = (const float*)d_in[17];

  const size_t SZ = (size_t)M_ROWS * HDIM;         // 50,331,648 bf16
  bf16* bufA = (bf16*)d_ws;                        // h  -> h2 -> g2out
  bf16* bufB = bufA + SZ;                          // q  -> temporal o
  bf16* bufC = bufB + SZ;                          // k  -> o@w_o out
  bf16* bufD = bufC + SZ;                          // hT -> g1out
  bf16* bufE = bufD + SZ;                          // qkv (M x 768)
  bf16* wb   = bufE + (size_t)M_ROWS * 768;
  bf16* ws1b  = wb;
  bf16* ws2b  = ws1b + 65536;
  bf16* wqkvb = ws2b + 65536;
  bf16* wob   = wqkvb + 196608;
  bf16* wg1b  = wob + 65536;
  bf16* wg2b  = wg1b + 65536;

  // weight casts to bf16
  k_cast<<<(65536 + 255) / 256, 256, 0, stream>>>(w_s1, ws1b, 65536);
  k_cast<<<(65536 + 255) / 256, 256, 0, stream>>>(w_s2, ws2b, 65536);
  k_cast<<<(196608 + 255) / 256, 256, 0, stream>>>(w_qkv, wqkvb, 196608);
  k_cast<<<(65536 + 255) / 256, 256, 0, stream>>>(w_o, wob, 65536);
  k_cast<<<(65536 + 255) / 256, 256, 0, stream>>>(w_g1, wg1b, 65536);
  k_cast<<<(65536 + 255) / 256, 256, 0, stream>>>(w_g2, wg2b, 65536);

  // h = x @ w_in.T + b_in  (bf16 out)
  k_in<<<M_ROWS, 256, 0, stream>>>(x, w_in, b_in, bufA);

  // q, k projections
  k_gemm<0, 0><<<dim3(M_ROWS / 64, 1), 256, LDSW_BYTES, stream>>>(bufA, ws1b, b_s1, bufB, 256);
  k_gemm<0, 0><<<dim3(M_ROWS / 64, 1), 256, LDSW_BYTES, stream>>>(bufA, ws2b, b_s2, bufC, 256);

  // hT for the P@V stage, then flash spatial attention: bufA <- h2
  k_transpose<<<M_ROWS, 256, 0, stream>>>(bufA, bufD);
  k_spattn<<<dim3(16, BT), 128, SPATTN_LDS, stream>>>(bufB, bufC, bufD, bufA);

  // qkv = ht @ w_qkv.T  (transpose folded into A-row remap)
  k_gemm<0, 1><<<dim3(M_ROWS / 64, 3), 256, LDSW_BYTES, stream>>>(bufA, wqkvb, b_qkv, bufE, 768);

  // temporal attention -> o (bufB), then o @ w_o.T -> bufC
  k_tattn<<<(BN * 8 * TSTEP) / 256, 256, 0, stream>>>(bufE, bufB);
  k_gemm<0, 0><<<dim3(M_ROWS / 64, 1), 256, LDSW_BYTES, stream>>>(bufB, wob, b_o, bufC, 256);

  // gate MLP with ReLU (layout transposes are no-ops for row-wise GEMMs)
  k_gemm<1, 0><<<dim3(M_ROWS / 64, 1), 256, LDSW_BYTES, stream>>>(bufC, wg1b, b_g1, bufD, 256);
  k_gemm<1, 0><<<dim3(M_ROWS / 64, 1), 256, LDSW_BYTES, stream>>>(bufD, wg2b, b_g2, bufA, 256);

  // final projection + scatter to (B,P,N,F)
  k_final<<<(BN * 36) / 256, 256, 0, stream>>>(bufA, w_out, b_out, (float*)d_out);
}
